// M_36472862278241
// MI455X (gfx1250) — matrix operand order repaired and checked by reading, not compiled, not run on hardware
//
#include <hip/hip_runtime.h>

typedef _Float16 half_t;
typedef __attribute__((ext_vector_type(16))) _Float16 v16h;
typedef __attribute__((ext_vector_type(8)))  _Float16 v8h;
typedef __attribute__((ext_vector_type(8)))  float    v8f;
typedef __attribute__((ext_vector_type(4)))  float    v4f;
typedef __attribute__((ext_vector_type(4)))  int      v4i;

#define CIN    672
#define COUT   128
#define HW     49
#define NTILE  64              // padded N (49 -> 64)
#define KT     32              // WMMA K per step (f16)
#define NKT    (CIN / KT)      // 21
#define XTILE  (KT * HW)       // 1568 floats, contiguous run of x per K-tile
#define BSTR   40              // sB row stride in halves (80B: 16B-aligned, non-pow2 banks)

#if __has_builtin(__builtin_amdgcn_global_load_async_to_lds_b128)
#define HAVE_ASYNC_LDS 1
typedef __attribute__((address_space(1))) v4i gas_v4i;   // global
typedef __attribute__((address_space(3))) v4i las_v4i;   // LDS
#endif

__device__ __forceinline__ void cp16_async(const float* g, float* l) {
#ifdef HAVE_ASYNC_LDS
    __builtin_amdgcn_global_load_async_to_lds_b128(
        (gas_v4i*)g, (las_v4i*)l, /*offset=*/0, /*cpol=*/0);
#else
    *(v4f*)l = *(const v4f*)g;      // sync fallback: vmem load + ds store
#endif
}

__device__ __forceinline__ void wait_async0() {
#if __has_builtin(__builtin_amdgcn_s_wait_asynccnt)
    __builtin_amdgcn_s_wait_asynccnt(0);
#else
    asm volatile("s_wait_asynccnt 0x0" ::: "memory");
#endif
}

// ---------------------------------------------------------------------------
// Prep: swizzle W (f32 [128][672]) into fragment-ready f16 in d_ws.
// Layout: for (kt, mtile, lane): 16 contiguous halves =
//   rows m = mtile*16 + (lane&15); K = kt*32 + akb + {0..7} and {16..23},
//   akb = (lane<16 ? 0 : 8)  — exactly the 16-bit A-fragment layout.
// ---------------------------------------------------------------------------
__global__ void w16_prep(const float* __restrict__ Wmat, half_t* __restrict__ W16) {
    int t = blockIdx.x * 256 + threadIdx.x;      // 0..5375 (21*8*32)
    int lane = t & 31;
    int mt   = (t >> 5) & 7;
    int kt   = t >> 8;
    int arow = mt * 16 + (lane & 15);
    int akb  = (lane < 16) ? 0 : 8;
    const float* src = Wmat + arow * CIN + kt * KT + akb;
    v4f a0 = *(const v4f*)(src);
    v4f a1 = *(const v4f*)(src + 4);
    v4f a2 = *(const v4f*)(src + 16);
    v4f a3 = *(const v4f*)(src + 20);
    v8h lo, hi;
    lo[0]=(half_t)a0.x; lo[1]=(half_t)a0.y; lo[2]=(half_t)a0.z; lo[3]=(half_t)a0.w;
    lo[4]=(half_t)a1.x; lo[5]=(half_t)a1.y; lo[6]=(half_t)a1.z; lo[7]=(half_t)a1.w;
    hi[0]=(half_t)a2.x; hi[1]=(half_t)a2.y; hi[2]=(half_t)a2.z; hi[3]=(half_t)a2.w;
    hi[4]=(half_t)a3.x; hi[5]=(half_t)a3.y; hi[6]=(half_t)a3.z; hi[7]=(half_t)a3.w;
    half_t* dst = W16 + (size_t)t * 16;
    *(v8h*)dst       = lo;
    *(v8h*)(dst + 8) = hi;
}

// ---------------------------------------------------------------------------
// Main: one block per batch image. M=128 (8 waves x 16), N=49->64, K=672.
// x K-tiles async-copied into double-buffered LDS; BN+ReLU+f16 pass writes
// sB [n][k]; WMMA f16 with f32 accumulators; A fragments stream from L2.
// ---------------------------------------------------------------------------
__global__ __launch_bounds__(256, 2)
void bn_gemm_fast(const float* __restrict__ x,
                  const float* __restrict__ gamma,
                  const float* __restrict__ beta,
                  const float* __restrict__ rmean,
                  const float* __restrict__ rvar,
                  const half_t* __restrict__ W16,
                  float* __restrict__ out)
{
    __shared__ float  sX[2][XTILE];          // raw x tiles (contiguous), 12.25 KB
    __shared__ float2 sSS[CIN];              // packed (scale, shift), 5.25 KB
    __shared__ half_t sB[NTILE * BSTR];      // BN'd f16 tile [n][k], 5 KB

    const int tid  = threadIdx.x;
    const int wave = tid >> 5;
    const int lane = tid & 31;
    const int b    = blockIdx.x;
    const float* xb = x + (size_t)b * CIN * HW;

    // BN fold: per-channel (scale, shift)
    for (int c = tid; c < CIN; c += 256) {
        float inv = rsqrtf(rvar[c] + 1e-5f);
        float s   = gamma[c] * inv;
        sSS[c] = make_float2(s, beta[c] - rmean[c] * s);
    }
    // zero sB once (pad rows n=49..63 stay zero forever)
    for (int i = tid; i < (NTILE * BSTR) / 2; i += 256) ((int*)sB)[i] = 0;

    // per-thread async-copy slots: 392 b128 transfers per tile (clamped -> uniform 2/thread)
    const int s0 = tid;
    const int s1 = (tid < XTILE / 4 - 256) ? tid + 256 : XTILE / 4 - 1;

    // per-thread BN slots (7 x 256 covers 1568 elems; flat index == c*49+n)
    int bnoff[7]; int bnsbo[7]; int bnc[7]; bool bnact[7];
    #pragma unroll
    for (int i = 0; i < 7; ++i) {
        int idx = tid + i * 256;
        bnact[i] = idx < XTILE;
        int idc  = bnact[i] ? idx : (XTILE - 1);
        int c = idc / HW, n = idc % HW;
        bnoff[i] = idc;                 // contiguous tile: flat offset
        bnsbo[i] = n * BSTR + c;
        bnc[i]   = c;
    }

    // accumulators + fragment addressing
    v8f acc[4];
    acc[0] = v8f{}; acc[1] = v8f{}; acc[2] = v8f{}; acc[3] = v8f{};
    const int bn  = lane & 15;
    const int bkb = (lane >> 4) * 8;
    const half_t* ap = W16 + wave * 512 + lane * 16;   // += 4096 halves per kt

    // prologue: async-stage tile 0
    cp16_async(xb + s0 * 4, &sX[0][s0 * 4]);
    cp16_async(xb + s1 * 4, &sX[0][s1 * 4]);
    wait_async0();
    __syncthreads();

    int buf = 0;
    for (int kt = 0; kt < NKT; ++kt) {
        const int k0 = kt * KT;

        // prefetch next tile into the other buffer (async, overlaps BN+WMMA)
        if (kt + 1 < NKT) {
            const float* gsrc = xb + (size_t)(kt + 1) * XTILE;
            cp16_async(gsrc + s0 * 4, &sX[buf ^ 1][s0 * 4]);
            cp16_async(gsrc + s1 * 4, &sX[buf ^ 1][s1 * 4]);
        }

        // BN + ReLU + f16 on current tile: sX[buf] -> sB
        const float* xr = sX[buf];
        #pragma unroll
        for (int i = 0; i < 7; ++i) {
            if (bnact[i]) {
                float2 ss = sSS[k0 + bnc[i]];
                float  v  = fmaxf(fmaf(xr[bnoff[i]], ss.x, ss.y), 0.0f);
                sB[bnsbo[i]] = (half_t)v;
            }
        }
        __syncthreads();                       // sB ready

        // A fragment: one 32B load from L2-hot swizzled W16
        v16h afrag = *(const v16h*)ap;
        ap += 4096;

        #pragma unroll
        for (int nt = 0; nt < 4; ++nt) {
            const half_t* bp = &sB[(nt * 16 + bn) * BSTR + bkb];
            v8h blo = *(const v8h*)bp;
            v8h bhi = *(const v8h*)(bp + 16);
            v16h bfrag = __builtin_shufflevector(blo, bhi,
                0,1,2,3,4,5,6,7,8,9,10,11,12,13,14,15);
            acc[nt] = __builtin_amdgcn_wmma_f32_16x16x32_f16(
                false, afrag, false, bfrag, (short)0, acc[nt], false, false);
        }

        wait_async0();                         // next tile resident (no-op on last)
        __syncthreads();                       // sB consumed; tile visible to all
        buf ^= 1;
    }

    // epilogue: C/D layout -> out[b][o][hw]
    float* outb = out + (size_t)b * COUT * HW;
    const int crow = wave * 16 + ((lane >> 4) * 8);
    const int ccol = lane & 15;
    #pragma unroll
    for (int nt = 0; nt < 4; ++nt) {
        int n = nt * 16 + ccol;
        if (n < HW) {
            #pragma unroll
            for (int r = 0; r < 8; ++r)
                outb[(size_t)(crow + r) * HW + n] = acc[nt][r];
        }
    }
}

// ---------------------------------------------------------------------------
// Fallback (no workspace): self-contained kernel.
// ---------------------------------------------------------------------------
__global__ __launch_bounds__(256, 2)
void bn_gemm_fallback(const float* __restrict__ x,
                      const float* __restrict__ gamma,
                      const float* __restrict__ beta,
                      const float* __restrict__ rmean,
                      const float* __restrict__ rvar,
                      const float* __restrict__ Wmat,
                      float* __restrict__ out)
{
    __shared__ float  sScale[CIN];
    __shared__ float  sShift[CIN];
    __shared__ half_t sA[COUT * KT];
    __shared__ half_t sB[NTILE * KT];

    const int tid  = threadIdx.x;
    const int wave = tid >> 5;
    const int lane = tid & 31;
    const int b    = blockIdx.x;
    const float* xb = x + (size_t)b * CIN * HW;

    for (int c = tid; c < CIN; c += 256) {
        float inv = rsqrtf(rvar[c] + 1e-5f);
        float s   = gamma[c] * inv;
        sScale[c] = s;
        sShift[c] = beta[c] - rmean[c] * s;
    }
    __syncthreads();

    v8f acc[4];
    acc[0] = v8f{}; acc[1] = v8f{}; acc[2] = v8f{}; acc[3] = v8f{};
    const int m0   = wave * 16;
    const int arow = m0 + (lane & 15);
    const int akb  = (lane < 16) ? 0 : 8;
    const int bn   = lane & 15;
    const int bkb  = (lane >> 4) * 8;

    for (int kt = 0; kt < NKT; ++kt) {
        const int k0 = kt * KT;
        #pragma unroll
        for (int i = 0; i < 4; ++i) {
            int idx = tid + i * 256, o = idx >> 3, cg = idx & 7;
            v4f wv = *(const v4f*)(Wmat + o * CIN + k0 + cg * 4);
            half_t* dst = &sA[o * KT + cg * 4];
            dst[0]=(half_t)wv.x; dst[1]=(half_t)wv.y; dst[2]=(half_t)wv.z; dst[3]=(half_t)wv.w;
        }
        #pragma unroll
        for (int i = 0; i < 8; ++i) {
            int idx = tid + i * 256, n = idx >> 5, c = idx & 31;
            float v = 0.0f;
            if (n < HW) {
                float xv = xb[(size_t)(k0 + c) * HW + n];
                v = fmaxf(fmaf(xv, sScale[k0 + c], sShift[k0 + c]), 0.0f);
            }
            sB[n * KT + c] = (half_t)v;
        }
        __syncthreads();

        v8h alo = *(const v8h*)&sA[arow * KT + akb];
        v8h ahi = *(const v8h*)&sA[arow * KT + akb + 16];
        v16h afrag = __builtin_shufflevector(alo, ahi,
            0,1,2,3,4,5,6,7,8,9,10,11,12,13,14,15);
        #pragma unroll
        for (int nt = 0; nt < 4; ++nt) {
            const half_t* bp = &sB[(nt * 16 + bn) * KT + bkb];
            v8h blo = *(const v8h*)bp;
            v8h bhi = *(const v8h*)(bp + 16);
            v16h bfrag = __builtin_shufflevector(blo, bhi,
                0,1,2,3,4,5,6,7,8,9,10,11,12,13,14,15);
            acc[nt] = __builtin_amdgcn_wmma_f32_16x16x32_f16(
                false, afrag, false, bfrag, (short)0, acc[nt], false, false);
        }
        __syncthreads();
    }

    float* outb = out + (size_t)b * COUT * HW;
    const int crow = m0 + ((lane >> 4) * 8);
    const int ccol = lane & 15;
    #pragma unroll
    for (int nt = 0; nt < 4; ++nt) {
        int n = nt * 16 + ccol;
        if (n < HW) {
            #pragma unroll
            for (int r = 0; r < 8; ++r)
                outb[(size_t)(crow + r) * HW + n] = acc[nt][r];
        }
    }
}

extern "C" void kernel_launch(void* const* d_in, const int* in_sizes, int n_in,
                              void* d_out, int out_size, void* d_ws, size_t ws_size,
                              hipStream_t stream) {
    const float* x     = (const float*)d_in[0];
    const float* gamma = (const float*)d_in[1];
    const float* beta  = (const float*)d_in[2];
    const float* rmean = (const float*)d_in[3];
    const float* rvar  = (const float*)d_in[4];
    const float* Wmat  = (const float*)d_in[5];
    float* out = (float*)d_out;
    (void)in_sizes; (void)n_in; (void)out_size;

    const size_t w16_bytes = (size_t)NKT * 8 * 32 * 16 * sizeof(half_t); // 172032
    if (ws_size >= w16_bytes) {
        half_t* W16 = (half_t*)d_ws;
        w16_prep<<<NKT, 256, 0, stream>>>(Wmat, W16);
        bn_gemm_fast<<<1024, 256, 0, stream>>>(x, gamma, beta, rmean, rvar, W16, out);
    } else {
        bn_gemm_fallback<<<1024, 256, 0, stream>>>(x, gamma, beta, rmean, rvar, Wmat, out);
    }
}
